// Discriminator_39900246180398
// MI455X (gfx1250) — matrix operand order repaired and checked by reading, not compiled, not run on hardware
//
#include <hip/hip_runtime.h>

// ---------------------------------------------------------------------------
// Types for CDNA5 WMMA (gfx1250): bf16 16x16x32, fp32 accum.
// ---------------------------------------------------------------------------
typedef __bf16 v16bf __attribute__((ext_vector_type(16)));
typedef float  v8f   __attribute__((ext_vector_type(8)));
typedef unsigned int v4u __attribute__((ext_vector_type(4)));

union BfFrag { v16bf v; v4u q[2]; };

#define WMMA_BF16(a, b, c) \
  __builtin_amdgcn_wmma_f32_16x16x32_bf16(false, (a), false, (b), (short)0, (c), false, false)

__device__ __forceinline__ unsigned short f2bf(float f) {
  unsigned u = __float_as_uint(f);
  unsigned r = u + 0x7fffu + ((u >> 16) & 1u);   // round-to-nearest-even
  return (unsigned short)(r >> 16);
}

// ---------------------------------------------------------------------------
// fp32 -> bf16 conversion with column padding (zeros)
// ---------------------------------------------------------------------------
__global__ void cvt_bf16_kernel(const float* __restrict__ src, unsigned short* __restrict__ dst,
                                long long rows, int cols, int colsPad) {
  long long i = (long long)blockIdx.x * blockDim.x + threadIdx.x;
  long long tot = rows * (long long)colsPad;
  if (i >= tot) return;
  int c = (int)(i % colsPad);
  long long r = i / colsPad;
  dst[i] = (c < cols) ? f2bf(src[r * cols + c]) : (unsigned short)0;
}

__global__ void fill_kernel(float* __restrict__ p, float v, long long n) {
  long long i = (long long)blockIdx.x * blockDim.x + threadIdx.x;
  if (i < n) p[i] = v;
}

// ---------------------------------------------------------------------------
// Generic WMMA GEMM: C[M,N] = act(A[M,K] @ W[N,K]^T + bias + resid)
// A, W bf16 row-major. Register-blocked: each wave computes a 32x64 output
// block (2 A-frags x 4 B-frags -> 8 WMMAs per 32-wide K step).
// Requires M % 32 == 0, N % 64 == 0, K % 32 == 0.  act: 0 = none, 1 = relu
// ---------------------------------------------------------------------------
__global__ void gemm_bf16_kernel(const unsigned short* __restrict__ A,
                                 const unsigned short* __restrict__ W,
                                 const float* __restrict__ bias,
                                 const float* __restrict__ resid,
                                 float* __restrict__ C,
                                 int M, int N, int K, int act) {
  int lane = threadIdx.x & 31;
  int wave = threadIdx.x >> 5;
  int nBlk = N >> 6;                      // 64-wide output blocks
  int mBlk = M >> 5;                      // 32-high output blocks
  int tile = blockIdx.x * 8 + wave;
  if (tile >= mBlk * nBlk) return;
  int mt = tile / nBlk, nt = tile % nBlk;
  int hiHalf = lane >> 4;

  const unsigned short* Arow0 = A + (size_t)((mt << 5) + (lane & 15)) * K + (hiHalf << 3);
  const unsigned short* Arow1 = Arow0 + (size_t)16 * K;
  const unsigned short* Wrow  = W + (size_t)((nt << 6) + (lane & 15)) * K + (hiHalf << 3);

  v8f acc[2][4] = {};
  for (int kk = 0; kk < K; kk += 32) {
    BfFrag a0, a1, b[4];
    a0.q[0] = *(const v4u*)(Arow0 + kk);        // K = k0 .. k0+7
    a0.q[1] = *(const v4u*)(Arow0 + kk + 16);   // K = k0+16 .. k0+23
    a1.q[0] = *(const v4u*)(Arow1 + kk);
    a1.q[1] = *(const v4u*)(Arow1 + kk + 16);
#pragma unroll
    for (int j = 0; j < 4; ++j) {
      const unsigned short* Wr = Wrow + (size_t)(j << 4) * K;
      b[j].q[0] = *(const v4u*)(Wr + kk);       // K = k0 .. k0+7
      b[j].q[1] = *(const v4u*)(Wr + kk + 16);  // K = k0+16 .. k0+23
    }
#pragma unroll
    for (int j = 0; j < 4; ++j) {
      acc[0][j] = WMMA_BF16(a0.v, b[j].v, acc[0][j]);
      acc[1][j] = WMMA_BF16(a1.v, b[j].v, acc[1][j]);
    }
  }

#pragma unroll
  for (int im = 0; im < 2; ++im) {
#pragma unroll
    for (int j = 0; j < 4; ++j) {
      int nCol = (nt << 6) + (j << 4) + (lane & 15);
      float bval = bias ? bias[nCol] : 0.f;
      int mBase = (mt << 5) + (im << 4) + (hiHalf << 3);
#pragma unroll
      for (int r = 0; r < 8; ++r) {
        size_t idx = (size_t)(mBase + r) * N + nCol;
        float v = acc[im][j][r] + bval;
        if (resid) v += resid[idx];
        if (act == 1) v = v > 0.f ? v : 0.f;
        C[idx] = v;
      }
    }
  }
}

// ---------------------------------------------------------------------------
// GATv2 edge kernels.  xl/xr: [n, 1024] (H=4 heads x C=256).  edge: [2,E].
// Edges e>=E are self loops (src=dst=e-E).
// ---------------------------------------------------------------------------
__global__ void gat_logit_kernel(const float* __restrict__ xl, const float* __restrict__ xr,
                                 const float* __restrict__ att, const int* __restrict__ edge,
                                 float* __restrict__ logit, float* __restrict__ mbuf,
                                 int E, int Etot) {
  int e = blockIdx.x;
  if (e >= Etot) return;
  int lane = threadIdx.x & 31, h = threadIdx.x >> 5;   // 128 threads: wave per head
  int s, d;
  if (e < E) { s = edge[e]; d = edge[E + e]; } else { s = d = e - E; }
  const float* pl = xl + (size_t)s * 1024 + h * 256;
  const float* pr = xr + (size_t)d * 1024 + h * 256;
  const float* pa = att + h * 256;
  float acc = 0.f;
  for (int c = lane; c < 256; c += 32) {
    float v = pl[c] + pr[c];
    v = v > 0.f ? v : 0.2f * v;                        // leaky relu
    acc += v * pa[c];
  }
  for (int o = 16; o; o >>= 1) acc += __shfl_xor(acc, o);
  if (lane == 0) {
    logit[e * 4 + h] = acc;
    float* addr = mbuf + d * 4 + h;
    if (acc >= 0.f) atomicMax((int*)addr, __float_as_int(acc));
    else            atomicMin((unsigned int*)addr, __float_as_uint(acc));
  }
}

__global__ void gat_expsum_kernel(const float* __restrict__ logit, const float* __restrict__ mbuf,
                                  const int* __restrict__ edge, float* __restrict__ exv,
                                  float* __restrict__ sbuf, int E, int Etot) {
  int i = blockIdx.x * blockDim.x + threadIdx.x;
  if (i >= Etot * 4) return;
  int e = i >> 2, h = i & 3;
  int d = (e < E) ? edge[E + e] : e - E;
  float ex = __expf(logit[i] - mbuf[d * 4 + h]);
  exv[i] = ex;
  atomicAdd(&sbuf[d * 4 + h], ex);
}

__global__ void gat_aggr_kernel(const float* __restrict__ xl, const float* __restrict__ exv,
                                const float* __restrict__ sbuf, const int* __restrict__ edge,
                                float* __restrict__ aggr, int E, int Etot) {
  int e = blockIdx.x;
  if (e >= Etot) return;
  int s, d;
  if (e < E) { s = edge[e]; d = edge[E + e]; } else { s = d = e - E; }
  for (int idx = threadIdx.x; idx < 1024; idx += 256) {
    int h = idx >> 8;
    float alpha = exv[e * 4 + h] / (sbuf[d * 4 + h] + 1e-16f);
    atomicAdd(&aggr[(size_t)d * 1024 + idx], alpha * xl[(size_t)s * 1024 + idx]);
  }
}

__global__ void gat_final_kernel(const float* __restrict__ aggr, const float* __restrict__ bias,
                                 float* __restrict__ yv, int n) {
  int i = blockIdx.x * blockDim.x + threadIdx.x;
  if (i >= n * 256) return;
  int node = i >> 8, c = i & 255;
  const float* a = aggr + (size_t)node * 1024;
  float s = (a[c] + a[256 + c] + a[512 + c] + a[768 + c]) * 0.25f + bias[c];
  yv[i] = s > 0.f ? s : __expf(s) - 1.f;               // elu
}

// ---------------------------------------------------------------------------
// QKV pack: qf [rows,128] f32, kvf [rows,256] f32 (k | v) ->
//   qpk [B,H,S,32] bf16 (scaled 1/sqrt(32)), kpk [B,H,S,32], vpk [B,H,32,S]
// ---------------------------------------------------------------------------
__global__ void pack_qkv_kernel(const float* __restrict__ qf, const float* __restrict__ kvf,
                                unsigned short* __restrict__ qpk, unsigned short* __restrict__ kpk,
                                unsigned short* __restrict__ vpk, int Bd, int S) {
  int i = blockIdx.x * blockDim.x + threadIdx.x;
  int tot = Bd * S * 128;
  if (i >= tot) return;
  int dIdx = i & 127;
  int s = (i >> 7) % S;
  int b = i / (128 * S);
  int h = dIdx >> 5, d = dIdx & 31;
  const float scale = 0.17677669529663689f;            // 1/sqrt(32)
  size_t row = (size_t)(b * S + s);
  float q  = qf[row * 128 + dIdx];
  float kk = kvf[row * 256 + dIdx];
  float vv = kvf[row * 256 + 128 + dIdx];
  size_t base = ((size_t)(b * 4 + h) * S + s) * 32 + d;
  qpk[base] = f2bf(q * scale);
  kpk[base] = f2bf(kk);
  vpk[((size_t)(b * 4 + h) * 32 + d) * S + s] = f2bf(vv);
}

// ---------------------------------------------------------------------------
// Fused attention: one block per (b, h, 32-query chunk).  Dynamic LDS:
//   scores fp32 [32,S] + P bf16 [32,S].  WMMA for QK^T and P@V.
// ---------------------------------------------------------------------------
__global__ void attn_kernel(const unsigned short* __restrict__ qpk,
                            const unsigned short* __restrict__ kpk,
                            const unsigned short* __restrict__ vpk,
                            float* __restrict__ O, int Bd, int S) {
  extern __shared__ char smem[];
  float* Ssc = (float*)smem;                                     // 32*S f32
  unsigned short* P = (unsigned short*)(smem + (size_t)32 * S * 4);
  int lane = threadIdx.x & 31, wave = threadIdx.x >> 5;
  int qChunks = S >> 5;
  int qc = blockIdx.x % qChunks;
  int h  = (blockIdx.x / qChunks) & 3;
  int b  = blockIdx.x / (qChunks * 4);
  int q0 = qc << 5;
  const unsigned short* Qb = qpk + (size_t)(b * 4 + h) * S * 32;
  const unsigned short* Kb = kpk + (size_t)(b * 4 + h) * S * 32;
  const unsigned short* Vb = vpk + (size_t)(b * 4 + h) * 32 * S;
  int hiHalf = lane >> 4;

  // Stage 1: scores = Q[32,32] @ K^T  (K-dim = hd = 32, single WMMA step)
  int nK = S >> 4;
  for (int t = wave; t < 2 * nK; t += 8) {
    int mt = t / nK, nt = t % nK;
    BfFrag a, bb;
    const unsigned short* Ar = Qb + (size_t)(q0 + (mt << 4) + (lane & 15)) * 32 + (hiHalf << 3);
    a.q[0] = *(const v4u*)(Ar);
    a.q[1] = *(const v4u*)(Ar + 16);
    const unsigned short* Br = Kb + (size_t)((nt << 4) + (lane & 15)) * 32 + (hiHalf << 3);
    bb.q[0] = *(const v4u*)(Br);
    bb.q[1] = *(const v4u*)(Br + 16);
    v8f c = {};
    c = WMMA_BF16(a.v, bb.v, c);
    int mBase = (mt << 4) + (hiHalf << 3);
    int col = (nt << 4) + (lane & 15);
#pragma unroll
    for (int r = 0; r < 8; ++r) Ssc[(mBase + r) * S + col] = c[r];
  }
  __syncthreads();

  // Stage 2: row softmax (wave per row)
  for (int row = wave; row < 32; row += 8) {
    float mx = -1e30f;
    for (int c = lane; c < S; c += 32) mx = fmaxf(mx, Ssc[row * S + c]);
    for (int o = 16; o; o >>= 1) mx = fmaxf(mx, __shfl_xor(mx, o));
    float sum = 0.f;
    for (int c = lane; c < S; c += 32) {
      float e = __expf(Ssc[row * S + c] - mx);
      Ssc[row * S + c] = e;
      sum += e;
    }
    for (int o = 16; o; o >>= 1) sum += __shfl_xor(sum, o);
    float inv = 1.f / sum;
    for (int c = lane; c < S; c += 32) P[row * S + c] = f2bf(Ssc[row * S + c] * inv);
  }
  __syncthreads();

  // Stage 3: O[32,32] = P[32,S] @ V[S,32]  (A-frags from LDS)
  if (wave < 4) {
    int mt = wave >> 1, nt = wave & 1;
    v8f acc = {};
    const unsigned short* Pr = P  + (size_t)((mt << 4) + (lane & 15)) * S + (hiHalf << 3);
    const unsigned short* Vr = Vb + (size_t)((nt << 4) + (lane & 15)) * S + (hiHalf << 3);
    for (int kk = 0; kk < S; kk += 32) {
      BfFrag a, bb;
      a.q[0] = *(const v4u*)(Pr + kk);
      a.q[1] = *(const v4u*)(Pr + kk + 16);
      bb.q[0] = *(const v4u*)(Vr + kk);
      bb.q[1] = *(const v4u*)(Vr + kk + 16);
      acc = WMMA_BF16(a.v, bb.v, acc);
    }
    int mBase = q0 + (mt << 4) + (hiHalf << 3);
    int col = h * 32 + (nt << 4) + (lane & 15);
#pragma unroll
    for (int r = 0; r < 8; ++r)
      O[(size_t)(b * S + mBase + r) * 128 + col] = acc[r];
  }
}

// ---------------------------------------------------------------------------
// LayerNorm over last dim 128; one block (128 threads) per row.
// ---------------------------------------------------------------------------
__global__ void ln_kernel(const float* __restrict__ x, const float* __restrict__ s,
                          const float* __restrict__ b, float* __restrict__ out) {
  __shared__ float red[128];
  int row = blockIdx.x, t = threadIdx.x;
  float v = x[(size_t)row * 128 + t];
  red[t] = v; __syncthreads();
  for (int o = 64; o; o >>= 1) { if (t < o) red[t] += red[t + o]; __syncthreads(); }
  float mu = red[0] * (1.f / 128.f);
  __syncthreads();
  float dv = v - mu;
  red[t] = dv * dv; __syncthreads();
  for (int o = 64; o; o >>= 1) { if (t < o) red[t] += red[t + o]; __syncthreads(); }
  float var = red[0] * (1.f / 128.f);
  out[(size_t)row * 128 + t] = dv * rsqrtf(var + 1e-5f) * s[t] + b[t];
}

// ---------------------------------------------------------------------------
// conv1 (1x1 over channel dim of reinterpreted [B,128,512]) + elu + avgpool4
// ---------------------------------------------------------------------------
__global__ void conv_pool_kernel(const float* __restrict__ Y, const float* __restrict__ w,
                                 const float* __restrict__ cb, float* __restrict__ z) {
  int i = blockIdx.x * blockDim.x + threadIdx.x;
  if (i >= 32 * 128) return;
  int b = i >> 7, d = i & 127;
  float bias = cb[0];
  float acc = 0.f;
  for (int j = 0; j < 4; ++j) {
    int t = d * 4 + j;
    float s = 0.f;
    for (int c = 0; c < 128; ++c) s += w[c] * Y[(size_t)b * 65536 + c * 512 + t];
    s += bias;
    acc += (s > 0.f ? s : __expf(s) - 1.f);
  }
  z[i] = acc * 0.25f;
}

// final: per-b dot(128) + tanh, mean over 32
__global__ void final_kernel(const float* __restrict__ X, const float* __restrict__ w2,
                             const float* __restrict__ b2, float* __restrict__ out) {
  int lane = threadIdx.x;   // 32 threads
  float s = 0.f;
  for (int c = 0; c < 128; ++c) s += w2[c] * X[lane * 128 + c];
  s = tanhf(s + b2[0]);
  for (int o = 16; o; o >>= 1) s += __shfl_xor(s, o);
  if (lane == 0) out[0] = s * (1.f / 32.f);
}

// ===========================================================================
// Host side
// ===========================================================================
namespace {

struct Ws {
  float *yv, *zbuf;
  unsigned short *xbg, *wlb, *wrb;
  float *xl, *xr, *logit, *exv, *mbuf, *sbuf, *aggr;
  unsigned short *memb, *xb, *wbuf, *qpk, *kpk, *vpk, *attnOb, *hidb;
  float *qf, *kvf, *attnO, *hid, *x2;
  int E, Etot;
};

inline void launch_cvt(const float* src, unsigned short* dst, long long rows, int cols,
                       int colsPad, hipStream_t st) {
  long long tot = rows * (long long)colsPad;
  cvt_bf16_kernel<<<(unsigned)((tot + 255) / 256), 256, 0, st>>>(src, dst, rows, cols, colsPad);
}

inline void launch_gemm(const unsigned short* A, const unsigned short* W, const float* bias,
                        const float* resid, float* C, int M, int N, int K, int act,
                        hipStream_t st) {
  int tiles = (M >> 5) * (N >> 6);
  gemm_bf16_kernel<<<(tiles + 7) / 8, 256, 0, st>>>(A, W, bias, resid, C, M, N, K, act);
}

void runGatLayer(const float* xin, int Kin, const float* Wl, const float* bl,
                 const float* Wr, const float* br, const float* att, const float* bout,
                 const int* edge, Ws& w, hipStream_t st) {
  const int n = 8192;
  int Kpad = (Kin + 31) & ~31;
  launch_cvt(xin, w.xbg, n, Kin, Kpad, st);
  launch_cvt(Wl, w.wlb, 1024, Kin, Kpad, st);
  launch_cvt(Wr, w.wrb, 1024, Kin, Kpad, st);
  launch_gemm(w.xbg, w.wlb, bl, nullptr, w.xl, n, 1024, Kpad, 0, st);
  launch_gemm(w.xbg, w.wrb, br, nullptr, w.xr, n, 1024, Kpad, 0, st);
  fill_kernel<<<(n * 4 + 255) / 256, 256, 0, st>>>(w.mbuf, -1e30f, n * 4);
  hipMemsetAsync(w.sbuf, 0, (size_t)n * 4 * 4, st);
  hipMemsetAsync(w.aggr, 0, (size_t)n * 1024 * 4, st);
  gat_logit_kernel<<<w.Etot, 128, 0, st>>>(w.xl, w.xr, att, edge, w.logit, w.mbuf, w.E, w.Etot);
  gat_expsum_kernel<<<(w.Etot * 4 + 255) / 256, 256, 0, st>>>(w.logit, w.mbuf, edge, w.exv,
                                                              w.sbuf, w.E, w.Etot);
  gat_aggr_kernel<<<w.Etot, 256, 0, st>>>(w.xl, w.exv, w.sbuf, edge, w.aggr, w.E, w.Etot);
  gat_final_kernel<<<(n * 256 + 255) / 256, 256, 0, st>>>(w.aggr, bout, w.yv, n);
}

void runMha(const unsigned short* qb, const unsigned short* kvb, const float* W,
            const float* bvec, const float* Wo, const float* bo, const float* resid,
            int Bd, int S, Ws& w, hipStream_t st) {
  int rows = Bd * S;
  launch_cvt(W, w.wbuf, 384, 128, 128, st);
  launch_gemm(qb,  w.wbuf,             bvec,       nullptr, w.qf,  rows, 128, 128, 0, st);
  launch_gemm(kvb, w.wbuf + 128 * 128, bvec + 128, nullptr, w.kvf, rows, 256, 128, 0, st);
  pack_qkv_kernel<<<(rows * 128 + 255) / 256, 256, 0, st>>>(w.qf, w.kvf, w.qpk, w.kpk, w.vpk,
                                                            Bd, S);
  size_t shmem = (size_t)32 * S * 4 + (size_t)32 * S * 2;
  attn_kernel<<<Bd * 4 * (S / 32), 256, shmem, st>>>(w.qpk, w.kpk, w.vpk, w.attnO, Bd, S);
  launch_cvt(w.attnO, w.attnOb, rows, 128, 128, st);
  launch_cvt(Wo, w.wbuf, 128, 128, 128, st);
  launch_gemm(w.attnOb, w.wbuf, bo, resid, w.x2, rows, 128, 128, 0, st);
}

void runDecoder(float* xf, int Bd, int S, void* const* d_in, int base, Ws& w, hipStream_t st) {
  int rows = Bd * S;
  const float* sa_w  = (const float*)d_in[base + 0];
  const float* sa_b  = (const float*)d_in[base + 1];
  const float* sa_ow = (const float*)d_in[base + 2];
  const float* sa_ob = (const float*)d_in[base + 3];
  const float* ca_w  = (const float*)d_in[base + 4];
  const float* ca_b  = (const float*)d_in[base + 5];
  const float* ca_ow = (const float*)d_in[base + 6];
  const float* ca_ob = (const float*)d_in[base + 7];
  const float* l1_w  = (const float*)d_in[base + 8];
  const float* l1_b  = (const float*)d_in[base + 9];
  const float* l2_w  = (const float*)d_in[base + 10];
  const float* l2_b  = (const float*)d_in[base + 11];
  const float* n1_s  = (const float*)d_in[base + 12];
  const float* n1_b  = (const float*)d_in[base + 13];
  const float* n2_s  = (const float*)d_in[base + 14];
  const float* n2_b  = (const float*)d_in[base + 15];
  const float* n3_s  = (const float*)d_in[base + 16];
  const float* n3_b  = (const float*)d_in[base + 17];

  launch_cvt(xf, w.memb, rows, 128, 128, st);   // mem stays the decoder input
  for (int i = 0; i < 2; ++i) {
    // self-attention
    launch_cvt(xf, w.xb, rows, 128, 128, st);
    runMha(w.xb, w.xb, sa_w + (size_t)i * 384 * 128, sa_b + i * 384,
           sa_ow + (size_t)i * 128 * 128, sa_ob + i * 128, xf, Bd, S, w, st);
    ln_kernel<<<rows, 128, 0, st>>>(w.x2, n1_s + i * 128, n1_b + i * 128, xf);
    // cross-attention (kv from mem)
    launch_cvt(xf, w.xb, rows, 128, 128, st);
    runMha(w.xb, w.memb, ca_w + (size_t)i * 384 * 128, ca_b + i * 384,
           ca_ow + (size_t)i * 128 * 128, ca_ob + i * 128, xf, Bd, S, w, st);
    ln_kernel<<<rows, 128, 0, st>>>(w.x2, n2_s + i * 128, n2_b + i * 128, xf);
    // FFN
    launch_cvt(xf, w.xb, rows, 128, 128, st);
    launch_cvt(l1_w + (size_t)i * 256 * 128, w.wbuf, 256, 128, 128, st);
    launch_gemm(w.xb, w.wbuf, l1_b + i * 256, nullptr, w.hid, rows, 256, 128, 1, st);
    launch_cvt(w.hid, w.hidb, rows, 256, 256, st);
    launch_cvt(l2_w + (size_t)i * 128 * 256, w.wbuf, 128, 256, 256, st);
    launch_gemm(w.hidb, w.wbuf, l2_b + i * 128, xf, w.x2, rows, 128, 256, 0, st);
    ln_kernel<<<rows, 128, 0, st>>>(w.x2, n3_s + i * 128, n3_b + i * 128, xf);
  }
}

} // namespace

extern "C" void kernel_launch(void* const* d_in, const int* in_sizes, int n_in,
                              void* d_out, int out_size, void* d_ws, size_t ws_size,
                              hipStream_t stream) {
  const float* to_score = (const float*)d_in[0];
  const int*   edge     = (const int*)d_in[1];
  const float* g0_Wl  = (const float*)d_in[2];
  const float* g0_bl  = (const float*)d_in[3];
  const float* g0_Wr  = (const float*)d_in[4];
  const float* g0_br  = (const float*)d_in[5];
  const float* g0_att = (const float*)d_in[6];
  const float* g0_bi  = (const float*)d_in[7];
  const float* g_Wl   = (const float*)d_in[8];
  const float* g_bl   = (const float*)d_in[9];
  const float* g_Wr   = (const float*)d_in[10];
  const float* g_br   = (const float*)d_in[11];
  const float* g_att  = (const float*)d_in[12];
  const float* g_bi   = (const float*)d_in[13];
  const float* conv1_w = (const float*)d_in[50];
  const float* conv1_b = (const float*)d_in[51];
  const float* conv2_w = (const float*)d_in[52];
  const float* conv2_b = (const float*)d_in[53];

  Ws w;
  w.E = in_sizes[1] / 2;          // 131072
  w.Etot = w.E + 8192;            // + self loops

  char* ws = (char*)d_ws;
  size_t off = 0;
  auto alloc = [&](size_t bytes) -> void* {
    void* p = ws + off;
    off = (off + bytes + 255) & ~(size_t)255;
    return p;
  };
  // persistent
  w.yv   = (float*)alloc((size_t)8192 * 256 * 4);
  w.zbuf = (float*)alloc((size_t)32 * 128 * 4);
  size_t shbase = off;
  // GAT stage scratch
  w.xbg   = (unsigned short*)alloc((size_t)8192 * 256 * 2);
  w.wlb   = (unsigned short*)alloc((size_t)1024 * 256 * 2);
  w.wrb   = (unsigned short*)alloc((size_t)1024 * 256 * 2);
  w.xl    = (float*)alloc((size_t)8192 * 1024 * 4);
  w.xr    = (float*)alloc((size_t)8192 * 1024 * 4);
  w.logit = (float*)alloc((size_t)w.Etot * 4 * 4);
  w.exv   = (float*)alloc((size_t)w.Etot * 4 * 4);
  w.mbuf  = (float*)alloc((size_t)8192 * 4 * 4);
  w.sbuf  = (float*)alloc((size_t)8192 * 4 * 4);
  w.aggr  = (float*)alloc((size_t)8192 * 1024 * 4);
  // decoder stage scratch (overlaps GAT scratch; stages are sequential)
  off = shbase;
  w.memb   = (unsigned short*)alloc((size_t)16384 * 128 * 2);
  w.xb     = (unsigned short*)alloc((size_t)16384 * 128 * 2);
  w.wbuf   = (unsigned short*)alloc((size_t)384 * 128 * 2);
  w.qf     = (float*)alloc((size_t)16384 * 128 * 4);
  w.kvf    = (float*)alloc((size_t)16384 * 256 * 4);
  w.qpk    = (unsigned short*)alloc((size_t)16384 * 128 * 2);
  w.kpk    = (unsigned short*)alloc((size_t)16384 * 128 * 2);
  w.vpk    = (unsigned short*)alloc((size_t)16384 * 128 * 2);
  w.attnO  = (float*)alloc((size_t)16384 * 128 * 4);
  w.attnOb = (unsigned short*)alloc((size_t)16384 * 128 * 2);
  w.hid    = (float*)alloc((size_t)16384 * 256 * 4);
  w.hidb   = (unsigned short*)alloc((size_t)16384 * 256 * 2);
  w.x2     = (float*)alloc((size_t)16384 * 128 * 4);

  // ---- GATv2 stack ----
  runGatLayer(to_score, 8, g0_Wl, g0_bl, g0_Wr, g0_br, g0_att, g0_bi, edge, w, stream);
  for (int i = 0; i < 3; ++i)
    runGatLayer(w.yv, 256, g_Wl + (size_t)i * 1024 * 256, g_bl + i * 1024,
                g_Wr + (size_t)i * 1024 * 256, g_br + i * 1024,
                g_att + i * 1024, g_bi + i * 256, edge, w, stream);

  // ---- decoder 1: [32,512,128] (reshape of yv) ----
  runDecoder(w.yv, 32, 512, d_in, 14, w, stream);

  // ---- conv1 + elu + avgpool -> z [32,128] ----
  conv_pool_kernel<<<(32 * 128 + 255) / 256, 256, 0, stream>>>(w.yv, conv1_w, conv1_b, w.zbuf);

  // ---- decoder 2: [1,32,128] ----
  runDecoder(w.zbuf, 1, 32, d_in, 32, w, stream);

  // ---- conv2 + tanh + mean -> scalar ----
  final_kernel<<<1, 32, 0, stream>>>(w.zbuf, conv2_w, conv2_b, (float*)d_out);
}
